// Mamba_layer_16999480558054
// MI455X (gfx1250) — matrix operand order repaired and checked by reading, not compiled, not run on hardware
//
#include <hip/hip_runtime.h>
#include <hip/hip_fp16.h>

typedef __attribute__((ext_vector_type(16))) _Float16 v16h;
typedef __attribute__((ext_vector_type(8)))  float    v8f;
typedef unsigned int uint4v __attribute__((ext_vector_type(4)));
typedef int          int8v  __attribute__((ext_vector_type(8)));
typedef int          int4v  __attribute__((ext_vector_type(4)));

#define BDIM  4
#define CINCH 64
#define LSEQ  4096
#define BL    (BDIM*LSEQ)   /* 16384 token rows */
#define DMM   256
#define DSTN  128
#define DINN  512
#define HDM   64
#define NHH   8
#define CONVD 768
#define DPROJ 1288

/* GEMM tiling: block 128x128, 8 waves (wave32), wave = 32x64 = 2x4 WMMA frags, BK=32 */
#define BMT 128
#define BNT 128
#define BKT 32
#define LDSF 36                       /* padded f32 row stride: 32 data + 4 pad DWORDs */
#define TILEF   (BMT * LDSF)          /* floats per tile      */
#define TILEBYT (TILEF * 4)           /* 18432 B              */
#define BUFBYT  (2 * TILEBYT)         /* A+B tile per buffer  */

/* ---- Tensor Data Mover: 2-D f32 tile load (tile 32 x 128) with LDS row padding ----
 * D# per CDNA5 ISA:
 *  group0: count=1 | lds_addr | global_addr[56:0] | type=2
 *  group1: data_size=2(4B), pad_enable, pad_interval=4 (32 DWORDs), pad_amount=3 (4 DWORDs),
 *          tensor_dim0/1 (zero-fill OOB), tile_dim0=32, tile_dim1=128, dim0 stride.      */
__device__ inline void tdm_load_2d_f32(unsigned lds_off, const float* gptr,
                                       unsigned tdim0, unsigned tdim1,
                                       unsigned long stride0)
{
    unsigned long ga = (unsigned long)(const void*)gptr;
    uint4v g0;
    g0[0] = 1u;                                        /* count=1, user mode         */
    g0[1] = lds_off;                                   /* LDS byte address           */
    g0[2] = (unsigned)(ga & 0xffffffffu);              /* global_addr[31:0]          */
    g0[3] = (unsigned)((ga >> 32) & 0x01ffffffu) | (2u << 30); /* addr[56:32]|type=2 */
    int8v g1;
    g1[0] = (int)((2u << 16) | (1u << 20) | (4u << 22) | (3u << 25));
    g1[1] = (int)((tdim0 & 0xffffu) << 16);                          /* tensor_dim0 lo */
    g1[2] = (int)(((tdim0 >> 16) & 0xffffu) | ((tdim1 & 0xffffu) << 16));
    g1[3] = (int)(((tdim1 >> 16) & 0xffffu) | ((unsigned)BKT << 16));/* tile_dim0=32   */
    g1[4] = (int)(unsigned)BMT;                                      /* tile_dim1=128  */
    g1[5] = (int)(unsigned)(stride0 & 0xffffffffu);                  /* dim0 stride lo */
    g1[6] = (int)(unsigned)((stride0 >> 32) & 0xffffu);              /* stride hi      */
    g1[7] = 0;
    int4v gz4 = {};
    int8v gz8 = {};
    __builtin_amdgcn_tensor_load_to_lds(g0, g1, gz4, gz4, gz8, 0);
}

__device__ inline v16h cvt16(float4 a, float4 b, float4 c, float4 d)
{
    v16h r;
    r[0]=(_Float16)a.x;  r[1]=(_Float16)a.y;  r[2]=(_Float16)a.z;  r[3]=(_Float16)a.w;
    r[4]=(_Float16)b.x;  r[5]=(_Float16)b.y;  r[6]=(_Float16)b.z;  r[7]=(_Float16)b.w;
    r[8]=(_Float16)c.x;  r[9]=(_Float16)c.y;  r[10]=(_Float16)c.z; r[11]=(_Float16)c.w;
    r[12]=(_Float16)d.x; r[13]=(_Float16)d.y; r[14]=(_Float16)d.z; r[15]=(_Float16)d.w;
    return r;
}

/* ---------------- WMMA GEMM with TDM double-buffered staging --------------------
 * C[M,N] = A[M,K] * W[N,K]^T (+bias); A row-major (lda), W row-major N x K.       */
__global__ __launch_bounds__(256)
void gemm_wmma_tdm(const float* __restrict__ A, int lda,
                   const float* __restrict__ W, const float* __restrict__ bias,
                   float* __restrict__ C, int ldc, int M, int N, int K)
{
    __shared__ __align__(16) float smem[2 * 2 * TILEF];   /* 2 buffers x (A,B) tiles */

    const int t    = threadIdx.x;
    const int lane = t & 31;
    const int wave = t >> 5;
    const int wm   = wave & 3;
    const int wn   = wave >> 2;
    const int m0   = blockIdx.x * BMT;
    const int n0   = blockIdx.y * BNT;
    const unsigned ldsbase = (unsigned)(size_t)(void*)smem;
    const int nk = K / BKT;

    v8f zero = {};
    v8f acc[2][4];
    #pragma unroll
    for (int i = 0; i < 2; ++i)
        #pragma unroll
        for (int j = 0; j < 4; ++j) acc[i][j] = zero;

    auto issue = [&](int kt) {
        const int k0 = kt * BKT;
        const unsigned boff = (unsigned)((kt & 1) * BUFBYT);
        tdm_load_2d_f32(ldsbase + boff,           A + (size_t)m0 * lda + k0,
                        (unsigned)(K - k0), (unsigned)(M - m0), (unsigned long)lda);
        tdm_load_2d_f32(ldsbase + boff + TILEBYT, W + (size_t)n0 * K + k0,
                        (unsigned)(K - k0), (unsigned)(N - n0), (unsigned long)K);
    };

    if (wave == 0) issue(0);

    const int fr    = lane & 15;
    const int koffA = (lane >> 4) * 8;   /* A: K in {koffA..+7} U {16+koffA..+7} */
    const int koffB = (lane >> 4) * 8;   /* B: K in {koffB..+7} U {16+koffB..+7} */

    for (int kt = 0; kt < nk; ++kt) {
        if (wave == 0) {
            if (kt + 1 < nk) {
                issue(kt + 1);
                __builtin_amdgcn_s_wait_tensorcnt(2);  /* current pair retired (in-order) */
            } else {
                __builtin_amdgcn_s_wait_tensorcnt(0);  /* last tile: drain                */
            }
        }
        __syncthreads();

        const float* Ab = smem + (kt & 1) * (2 * TILEF);
        const float* Bb = Ab + TILEF;

        v16h af[2], bf[4];
        #pragma unroll
        for (int i = 0; i < 2; ++i) {
            const float* p = Ab + (wm * 32 + i * 16 + fr) * LDSF;
            af[i] = cvt16(*(const float4*)(p + koffA),      *(const float4*)(p + koffA + 4),
                          *(const float4*)(p + 16 + koffA), *(const float4*)(p + 20 + koffA));
        }
        #pragma unroll
        for (int j = 0; j < 4; ++j) {
            const float* p = Bb + (wn * 64 + j * 16 + fr) * LDSF;
            bf[j] = cvt16(*(const float4*)(p + koffB),      *(const float4*)(p + koffB + 4),
                          *(const float4*)(p + 16 + koffB), *(const float4*)(p + 20 + koffB));
        }
        #pragma unroll
        for (int i = 0; i < 2; ++i)
            #pragma unroll
            for (int j = 0; j < 4; ++j)
                acc[i][j] = __builtin_amdgcn_wmma_f32_16x16x32_f16(
                    false, af[i], false, bf[j], (short)0, acc[i][j], false, false);
        __syncthreads();
    }

    /* D layout: m = v + 8*(lane>>4), n = lane&15 */
    const int half = lane >> 4;
    #pragma unroll
    for (int j = 0; j < 4; ++j) {
        int ng = n0 + wn * 64 + j * 16 + fr;
        if (ng >= N) continue;
        float bv = bias ? bias[ng] : 0.0f;
        #pragma unroll
        for (int i = 0; i < 2; ++i) {
            #pragma unroll
            for (int v = 0; v < 8; ++v) {
                int mg = m0 + wm * 32 + i * 16 + half * 8 + v;
                C[(size_t)mg * ldc + ng] = acc[i][j][v] + bv;
            }
        }
    }
}

/* ---------------- x (B,CIN,L) -> xt (B*L, CIN), LDS-tiled transpose ------------ */
__global__ __launch_bounds__(256)
void transpose_x_kernel(const float* __restrict__ x, float* __restrict__ xt)
{
    __shared__ float tile[32][33];
    const int lb = blockIdx.x * 32;           /* 128 blocks along L  */
    const int b  = (int)blockIdx.y >> 1;      /* 8 blocks: B x 2 c-tiles */
    const int cb = ((int)blockIdx.y & 1) * 32;
    const int tx = threadIdx.x & 31;
    const int ty = threadIdx.x >> 5;          /* 0..7 */
    for (int i = ty; i < 32; i += 8)
        tile[i][tx] = x[((size_t)b * CINCH + cb + i) * LSEQ + lb + tx];
    __syncthreads();
    for (int i = ty; i < 32; i += 8)
        xt[((size_t)b * LSEQ + lb + i) * CINCH + cb + tx] = tile[tx][i];
}

/* ---------------- dt = softplus(z[...,1280+h] + dt_bias); dA = exp(dt*A) ------- */
__global__ void dtprep_kernel(const float* __restrict__ z,
                              const float* __restrict__ dt_bias,
                              const float* __restrict__ A_log,
                              float* __restrict__ dt, float* __restrict__ da)
{
    int idx = blockIdx.x * blockDim.x + threadIdx.x;
    if (idx >= BL * NHH) return;
    int h = idx & 7;
    size_t m = (size_t)(idx >> 3);
    float v  = z[m * DPROJ + (DINN + CONVD) + h] + dt_bias[h];
    float sp = (v > 20.0f) ? v : log1pf(expf(v));
    float Av = -expf(A_log[h]);
    dt[idx] = sp;
    da[idx] = expf(sp * Av);
}

/* ---------------- depthwise conv (k=4, causal) + silu over xBC ---------------- */
__global__ void conv_kernel(const float* __restrict__ z,
                            const float* __restrict__ cw,
                            const float* __restrict__ cb,
                            float* __restrict__ xc)
{
    size_t idx = (size_t)blockIdx.x * blockDim.x + threadIdx.x;
    if (idx >= (size_t)BL * CONVD) return;
    int    j = (int)(idx % CONVD);
    size_t m = idx / CONVD;
    int    l = (int)(m & (LSEQ - 1));
    float s = cb[j];
    #pragma unroll
    for (int k = 0; k < 4; ++k) {
        int ls = l + k - 3;
        if (ls >= 0)
            s += z[(m + (size_t)(k - 3)) * DPROJ + DINN + j] * cw[j * 4 + k];
    }
    xc[m * CONVD + j] = s / (1.0f + expf(-s));   /* silu */
}

/* ---------------- sequential SSM scan; 1 block per (b,h) --------------------- */
__global__ __launch_bounds__(256)
void scan_kernel(const float* __restrict__ xc,  /* [BL][CONVD] */
                 const float* __restrict__ dtb, /* [BL][NH]    */
                 const float* __restrict__ dab, /* [BL][NH]    */
                 float* __restrict__ y)         /* [BL][DIN]   */
{
    const int b  = blockIdx.x >> 3;
    const int h  = blockIdx.x & 7;
    const int t  = threadIdx.x;
    const int p  = t & 63;
    const int ng = t >> 6;
    __shared__ float xs[HDM];
    __shared__ float Bsh[DSTN];
    __shared__ float Csh[DSTN];
    __shared__ float ys[4 * HDM];

    float hst[32];
    #pragma unroll
    for (int j = 0; j < 32; ++j) hst[j] = 0.0f;

    for (int l = 0; l < LSEQ; ++l) {
        const size_t m  = (size_t)b * LSEQ + l;
        const float* row = xc + m * CONVD;
        for (int i = t; i < HDM + 2 * DSTN; i += 256) {
            if (i < HDM)              xs[i]               = row[h * HDM + i];
            else if (i < HDM + DSTN)  Bsh[i - HDM]        = row[DINN + (i - HDM)];
            else                      Csh[i - HDM - DSTN] = row[DINN + DSTN + (i - HDM - DSTN)];
            if (l + 1 < LSEQ) __builtin_prefetch(row + CONVD + i, 0, 0);
        }
        float dtv = dtb[m * NHH + h];
        float dav = dab[m * NHH + h];
        __syncthreads();

        float cxp = dtv * xs[p];
        float acc = 0.0f;
        const int nb = ng * 32;
        #pragma unroll
        for (int j = 0; j < 32; ++j) {
            float hv = fmaf(hst[j], dav, cxp * Bsh[nb + j]);
            hst[j] = hv;
            acc = fmaf(hv, Csh[nb + j], acc);
        }
        ys[ng * HDM + p] = acc;
        __syncthreads();
        if (ng == 0)
            y[m * DINN + h * HDM + p] = ys[p] + ys[HDM + p] + ys[2 * HDM + p] + ys[3 * HDM + p];
        __syncthreads();
    }
}

/* ---------------- y += Dh*xh ; y *= silu(z) ; RMSNorm (in-place) -------------- */
__global__ __launch_bounds__(256)
void gatenorm_kernel(float* __restrict__ y, const float* __restrict__ xc,
                     const float* __restrict__ z,
                     const float* __restrict__ Dh, const float* __restrict__ nw)
{
    const size_t m = blockIdx.x;
    const int t = threadIdx.x;
    __shared__ float red[256];
    float vv[2];
    float ss = 0.0f;
    #pragma unroll
    for (int u = 0; u < 2; ++u) {
        int i = t + u * 256;
        float yv = y[m * DINN + i] + Dh[i >> 6] * xc[m * CONVD + i];
        float zv = z[m * DPROJ + i];
        yv *= zv / (1.0f + expf(-zv));
        vv[u] = yv;
        ss += yv * yv;
    }
    red[t] = ss;
    __syncthreads();
    for (int s = 128; s > 0; s >>= 1) {
        if (t < s) red[t] += red[t + s];
        __syncthreads();
    }
    float rms = rsqrtf(red[0] / (float)DINN + 1e-5f);
    #pragma unroll
    for (int u = 0; u < 2; ++u) {
        int i = t + u * 256;
        y[m * DINN + i] = vv[u] * rms * nw[i];
    }
}

/* ---------------- g = x_s * sigmoid(x_mo) (in-place over x_s) ----------------- */
__global__ void gmul_kernel(float* __restrict__ xs, const float* __restrict__ xm)
{
    size_t idx = (size_t)blockIdx.x * blockDim.x + threadIdx.x;
    if (idx >= (size_t)BL * DMM) return;
    xs[idx] = xs[idx] * (1.0f / (1.0f + expf(-xm[idx])));
}

/* ---------------- out[b][c][l] = o1[m][c] + res[m][c] ------------------------- */
__global__ void out_kernel(const float* __restrict__ o1, const float* __restrict__ res,
                           float* __restrict__ out)
{
    size_t idx = (size_t)blockIdx.x * blockDim.x + threadIdx.x;
    if (idx >= (size_t)BL * CINCH) return;
    int    c = (int)(idx & 63);
    size_t m = idx >> 6;
    int    b = (int)(m >> 12);
    int    l = (int)(m & (LSEQ - 1));
    out[((size_t)b * CINCH + c) * LSEQ + l] = o1[idx] + res[idx];
}

extern "C" void kernel_launch(void* const* d_in, const int* in_sizes, int n_in,
                              void* d_out, int out_size, void* d_ws, size_t ws_size,
                              hipStream_t stream)
{
    const float* x        = (const float*)d_in[0];
    const float* conv1d_w = (const float*)d_in[1];
    const float* W1       = (const float*)d_in[2];
    const float* b1       = (const float*)d_in[3];
    const float* W2       = (const float*)d_in[4];
    const float* b2       = (const float*)d_in[5];
    const float* W3       = (const float*)d_in[6];
    const float* b3       = (const float*)d_in[7];
    const float* Win      = (const float*)d_in[8];
    const float* convw    = (const float*)d_in[9];
    const float* convb    = (const float*)d_in[10];
    const float* dt_bias  = (const float*)d_in[11];
    const float* A_log    = (const float*)d_in[12];
    const float* Dh       = (const float*)d_in[13];
    const float* norm_w   = (const float*)d_in[14];
    const float* Wout     = (const float*)d_in[15];
    float* out = (float*)d_out;

    float* ws = (float*)d_ws;
    size_t o = 0;
    float* Z   = ws + o; o += (size_t)BL * DPROJ;  /* zxbcdt          */
    float* XT  = ws + o; o += (size_t)BL * CINCH;  /* x transposed    */
    float* XM  = ws + o; o += (size_t)BL * DMM;    /* x_m, then x_mo  */
    float* XS  = ws + o; o += (size_t)BL * DMM;    /* x_s, then g     */
    float* XC  = ws + o; o += (size_t)BL * CONVD;  /* conv+silu xBC   */
    float* DT  = ws + o; o += (size_t)BL * NHH;
    float* DA  = ws + o; o += (size_t)BL * NHH;
    float* Y   = ws + o; o += (size_t)BL * DINN;   /* scan y, then yn */
    float* O1  = ws + o; o += (size_t)BL * CINCH;
    float* RES = ws + o; o += (size_t)BL * CINCH;
    (void)ws_size; (void)in_sizes; (void)n_in; (void)out_size;

    dim3 blk(256);
    auto gemm = [&](const float* A, int lda, const float* W,
                    const float* bias, float* C, int ldc, int M, int N, int K) {
        dim3 g(M / BMT, (N + BNT - 1) / BNT);
        gemm_wmma_tdm<<<g, blk, 0, stream>>>(A, lda, W, bias, C, ldc, M, N, K);
    };

    transpose_x_kernel<<<dim3(LSEQ / 32, BDIM * 2), blk, 0, stream>>>(x, XT);
    gemm(XT, CINCH, W1, b1, XM, DMM, BL, DMM, CINCH);          /* x_m    */
    gemm(XT, CINCH, W2, b2, XS, DMM, BL, DMM, CINCH);          /* x_s    */
    gemm(XM, DMM, Win, nullptr, Z, DPROJ, BL, DPROJ, DMM);     /* zxbcdt */
    dtprep_kernel<<<(BL * NHH + 255) / 256, blk, 0, stream>>>(Z, dt_bias, A_log, DT, DA);
    conv_kernel<<<(int)(((size_t)BL * CONVD + 255) / 256), blk, 0, stream>>>(Z, convw, convb, XC);
    scan_kernel<<<BDIM * NHH, blk, 0, stream>>>(XC, DT, DA, Y);
    gatenorm_kernel<<<BL, blk, 0, stream>>>(Y, XC, Z, Dh, norm_w);
    gemm(Y, DINN, Wout, nullptr, XM, DMM, BL, DMM, DINN);      /* x_mo   */
    gmul_kernel<<<(BL * DMM + 255) / 256, blk, 0, stream>>>(XS, XM);
    gemm(XS, DMM, W3, b3, O1, CINCH, BL, CINCH, DMM);          /* o1     */
    gemm(XT, CINCH, conv1d_w, nullptr, RES, CINCH, BL, CINCH, CINCH); /* residual */
    out_kernel<<<(BL * CINCH + 255) / 256, blk, 0, stream>>>(O1, RES, out);
}
